// TIPSv2TextAttention_31044023615550
// MI455X (gfx1250) — compile-verified, matrix operand order checked by reading, not run on hardware
//
#include <hip/hip_runtime.h>
#include <hip/hip_bf16.h>

// ---------------------------------------------------------------------------
// MHA for B=4, S=2048, D=1024, H=16, HD=64 on gfx1250 (wave32, WMMA bf16)
// ---------------------------------------------------------------------------

#define Bq 4
#define Sq 2048
#define Dq 1024
#define Hq 16
#define HDq 64

typedef __attribute__((ext_vector_type(16))) __bf16 v16bf;
typedef __attribute__((ext_vector_type(8)))  float  v8f;
typedef unsigned short u16_t;

__device__ __forceinline__ u16_t f2bf(float f) {
    unsigned u = __builtin_bit_cast(unsigned, f);
    unsigned r = u + 0x7FFFu + ((u >> 16) & 1u);   // round-to-nearest-even
    return (u16_t)(r >> 16);
}

__device__ __forceinline__ v8f wmma_bf16(v16bf a, v16bf b, v8f c) {
    return __builtin_amdgcn_wmma_f32_16x16x32_bf16(
        false, a, false, b, (short)0, c, false, false);
}

// Load a 16x32 bf16 WMMA fragment (A or B role) from a row-major bf16 matrix.
// "row" dim = (lane&15)+row0, contiguous dim = k.  Per ISA 7.12.2:
//   lanes 0-15  : pairs cover k = {0,1..6,7, 16,17..22,23} (+k0)
//   lanes 16-31 : pairs cover k = {8,9..14,15, 24,25..30,31} (+k0)
__device__ __forceinline__ v16bf load_frag(const u16_t* base, int row0, int ld, int k0) {
    const int lane = threadIdx.x & 31;
    const int r    = lane & 15;
    const int hb   = (lane >> 4) * 4;      // uint offset for lane-half K split
    union { v16bf v; unsigned u[8]; } f;
    const unsigned* rp = (const unsigned*)(base + (size_t)(row0 + r) * ld + k0);
#pragma unroll
    for (int j = 0; j < 4; ++j) f.u[j] = rp[j + hb];
#pragma unroll
    for (int j = 0; j < 4; ++j) f.u[4 + j] = rp[8 + j + hb];
    return f.v;
}

// ---------------------------------------------------------------------------
// 1) fp32 -> bf16 conversions
// ---------------------------------------------------------------------------
__global__ __launch_bounds__(256) void cvt_bf16_kernel(const float* __restrict__ in,
                                                       u16_t* __restrict__ out, int n) {
    int i = blockIdx.x * 256 + threadIdx.x;
    if (i < n) out[i] = f2bf(in[i]);
}

// in: K x N row-major fp32  ->  out: N x K row-major bf16 (transposed)
__global__ __launch_bounds__(256) void transpose_bf16_kernel(const float* __restrict__ in,
                                                             u16_t* __restrict__ out,
                                                             int K, int N) {
    int i = blockIdx.x * 256 + threadIdx.x;
    if (i < K * N) {
        int n = i / K, k = i % K;
        out[i] = f2bf(in[(size_t)k * N + n]);
    }
}

// ---------------------------------------------------------------------------
// Double-buffered 32x64 GEMM core: acc[2][4] += A(32xK) * B(Kx64)
// abase rows row0..row0+31 ; bbase rows bcol0..bcol0+63 (both ld = Dq = K)
// ---------------------------------------------------------------------------
struct Tile {
    v16bf a[2];
    v16bf b[4];
};

__device__ __forceinline__ void load_tile(Tile& t, const u16_t* abase, int row0,
                                          const u16_t* bbase, int bcol0, int k0) {
#pragma unroll
    for (int m = 0; m < 2; ++m) t.a[m] = load_frag(abase, row0 + m * 16, Dq, k0);
#pragma unroll
    for (int c = 0; c < 4; ++c) t.b[c] = load_frag(bbase, bcol0 + c * 16, Dq, k0);
}

__device__ __forceinline__ void mma_tile(v8f acc[2][4], const Tile& t) {
#pragma unroll
    for (int m = 0; m < 2; ++m)
#pragma unroll
        for (int c = 0; c < 4; ++c)
            acc[m][c] = wmma_bf16(t.a[m], t.b[c], acc[m][c]);
}

__device__ __forceinline__ void gemm_core(v8f acc[2][4], const u16_t* abase, int row0,
                                          const u16_t* bbase, int bcol0) {
    Tile t0, t1;
    load_tile(t0, abase, row0, bbase, bcol0, 0);          // prologue: k=0
    for (int k0 = 0; k0 < Dq; k0 += 64) {
        load_tile(t1, abase, row0, bbase, bcol0, k0 + 32); // prefetch k0+32
        mma_tile(acc, t0);                                 // compute k0
        if (k0 + 64 < Dq)
            load_tile(t0, abase, row0, bbase, bcol0, k0 + 64); // prefetch k0+64
        mma_tile(acc, t1);                                 // compute k0+32
    }
}

// ---------------------------------------------------------------------------
// 2) QKV GEMM: (B*S,1024) x (1024,3072) + bias.  Writes Q(scaled),K as
//    (B,H,S,64) bf16 and V transposed as (B,H,64,S) bf16.
// ---------------------------------------------------------------------------
__global__ __launch_bounds__(256) void qkv_gemm_kernel(const u16_t* __restrict__ xb,
                                                       const u16_t* __restrict__ wiT,
                                                       const float* __restrict__ b_in,
                                                       u16_t* __restrict__ Qb,
                                                       u16_t* __restrict__ Kb,
                                                       u16_t* __restrict__ Vt) {
    const int wave = threadIdx.x >> 5;
    const int lane = threadIdx.x & 31;
    const int task = blockIdx.x * 8 + wave;          // 256 mtiles * 48 ngroups
    const int mtile = task / 48;
    const int ngrp  = task % 48;
    const int row0  = mtile * 32;

    v8f acc[2][4] = {};
    gemm_core(acc, xb, row0, wiT, ngrp * 64);

    const int half = lane >> 4;
    const int n    = lane & 15;
#pragma unroll
    for (int c = 0; c < 4; ++c) {
        int col   = ngrp * 64 + c * 16 + n;
        int which = col >> 10;           // 0=Q 1=K 2=V
        int d     = col & 1023;
        int h     = d >> 6;
        int hd    = d & 63;
        float bias = b_in[col];
#pragma unroll
        for (int m = 0; m < 2; ++m)
#pragma unroll
            for (int r = 0; r < 8; ++r) {
                int grow = row0 + m * 16 + r + 8 * half;
                int b    = grow >> 11;
                int s    = grow & 2047;
                float v  = acc[m][c][r] + bias;
                size_t bh = (size_t)(b * Hq + h);
                if (which == 0) {
                    Qb[(bh * Sq + s) * HDq + hd] = f2bf(v * 0.125f);  // 1/sqrt(64)
                } else if (which == 1) {
                    Kb[(bh * Sq + s) * HDq + hd] = f2bf(v);
                } else {
                    Vt[(bh * HDq + hd) * Sq + s] = f2bf(v);
                }
            }
    }
}

// ---------------------------------------------------------------------------
// 3) Flash attention: per wave one 16-row q tile of one (b,h).
//    Scores computed transposed: S^T = K_tile(16x64) x Q^T(64x16).
// ---------------------------------------------------------------------------
__global__ __launch_bounds__(256) void flash_attn_kernel(const u16_t* __restrict__ Qb,
                                                         const u16_t* __restrict__ Kb,
                                                         const u16_t* __restrict__ Vt,
                                                         const int* __restrict__ mask,
                                                         u16_t* __restrict__ attn_out) {
    __shared__ int   smask[Sq];
    __shared__ u16_t sP[8][16 * 32];        // per-wave P staging tile

    const int wave = threadIdx.x >> 5;
    const int lane = threadIdx.x & 31;
    const int half = lane >> 4;
    const int n    = lane & 15;

    const int bh = blockIdx.x >> 4;         // 16 q-blocks of 128 rows per (b,h)
    const int qb = blockIdx.x & 15;
    const int b  = bh >> 4;
    const int h  = bh & 15;
    const int q0 = qb * 128 + wave * 16;

    for (int i = threadIdx.x; i < Sq; i += 256) smask[i] = mask[b * Sq + i];
    __syncthreads();

    const u16_t* Qh = Qb + (size_t)bh * Sq * HDq;
    const u16_t* Kh = Kb + (size_t)bh * Sq * HDq;
    const u16_t* Vh = Vt + (size_t)bh * HDq * Sq;

    v16bf qf0 = load_frag(Qh, q0, HDq, 0);
    v16bf qf1 = load_frag(Qh, q0, HDq, 32);

    float mrun = -1e30f, lrun = 0.0f;
    v8f o[4] = {};

    for (int kv0 = 0; kv0 < Sq; kv0 += 32) {
        v8f s[2];
#pragma unroll
        for (int g = 0; g < 2; ++g) {
            v8f sg = {};
            v16bf a0 = load_frag(Kh, kv0 + 16 * g, HDq, 0);
            v16bf a1 = load_frag(Kh, kv0 + 16 * g, HDq, 32);
            sg = wmma_bf16(a0, qf0, sg);
            sg = wmma_bf16(a1, qf1, sg);
#pragma unroll
            for (int r = 0; r < 8; ++r) {
                int key = kv0 + 16 * g + r + 8 * half;     // C-frag M = key
                if (smask[key] == 0) sg[r] = -1e30f;
            }
            s[g] = sg;
        }
        // online softmax over the 32-key tile; stats live per q-column n
        float tm = -1e30f;
#pragma unroll
        for (int g = 0; g < 2; ++g)
#pragma unroll
            for (int r = 0; r < 8; ++r) tm = fmaxf(tm, s[g][r]);
        tm = fmaxf(tm, __shfl_xor(tm, 16, 32));
        float newm  = fmaxf(mrun, tm);
        float alpha = __expf(mrun - newm);
        mrun = newm;

        float rsum = 0.0f;
#pragma unroll
        for (int g = 0; g < 2; ++g)
#pragma unroll
            for (int r = 0; r < 8; ++r) {
                float p = __expf(s[g][r] - newm);
                s[g][r] = p;
                rsum += p;
            }
        rsum += __shfl_xor(rsum, 16, 32);
        lrun = lrun * alpha + rsum;

        // stage P (bf16) into LDS as a 16(q) x 32(key) tile
#pragma unroll
        for (int g = 0; g < 2; ++g)
#pragma unroll
            for (int r = 0; r < 8; ++r)
                sP[wave][n * 32 + g * 16 + r + 8 * half] = f2bf(s[g][r]);
        asm volatile("s_wait_dscnt 0" ::: "memory");

        v16bf pf = load_frag(&sP[wave][0], 0, 32, 0);      // A-frag: M=q, K=key

        float a8[8];
#pragma unroll
        for (int r = 0; r < 8; ++r) a8[r] = __shfl(alpha, r + 8 * half, 32);

#pragma unroll
        for (int c = 0; c < 4; ++c) {
            v16bf vf = load_frag(Vh, c * 16, Sq, kv0);     // B-frag: K=key, N=hd
#pragma unroll
            for (int r = 0; r < 8; ++r) o[c][r] *= a8[r];  // rescale by row M
            o[c] = wmma_bf16(pf, vf, o[c]);
        }
    }

    float li[8];
#pragma unroll
    for (int r = 0; r < 8; ++r) {
        float l = __shfl(lrun, r + 8 * half, 32);
        li[r] = (l > 0.0f) ? 1.0f / l : 0.0f;
    }
#pragma unroll
    for (int c = 0; c < 4; ++c)
#pragma unroll
        for (int r = 0; r < 8; ++r) {
            int row = q0 + r + 8 * half;
            int col = h * HDq + c * 16 + n;
            attn_out[((size_t)b * Sq + row) * Dq + col] = f2bf(o[c][r] * li[r]);
        }
}

// ---------------------------------------------------------------------------
// 4) Output GEMM: (B*S,1024) x (1024,1024) + bias -> fp32 d_out
// ---------------------------------------------------------------------------
__global__ __launch_bounds__(256) void out_gemm_kernel(const u16_t* __restrict__ ab,
                                                       const u16_t* __restrict__ woT,
                                                       const float* __restrict__ b_out,
                                                       float* __restrict__ out) {
    const int wave = threadIdx.x >> 5;
    const int lane = threadIdx.x & 31;
    const int task = blockIdx.x * 8 + wave;          // 256 mtiles * 16 ngroups
    const int mtile = task >> 4;
    const int ngrp  = task & 15;
    const int row0  = mtile * 32;

    v8f acc[2][4] = {};
    gemm_core(acc, ab, row0, woT, ngrp * 64);

    const int half = lane >> 4;
    const int n    = lane & 15;
#pragma unroll
    for (int c = 0; c < 4; ++c) {
        int col = ngrp * 64 + c * 16 + n;
        float bias = b_out[col];
#pragma unroll
        for (int m = 0; m < 2; ++m)
#pragma unroll
            for (int r = 0; r < 8; ++r) {
                int row = row0 + m * 16 + r + 8 * half;
                out[(size_t)row * Dq + col] = acc[m][c][r] + bias;
            }
    }
}

// ---------------------------------------------------------------------------
extern "C" void kernel_launch(void* const* d_in, const int* in_sizes, int n_in,
                              void* d_out, int out_size, void* d_ws, size_t ws_size,
                              hipStream_t stream) {
    const float* x     = (const float*)d_in[0];   // (B,S,D)
    const float* w_in  = (const float*)d_in[1];   // (D, 3D)
    const float* b_in  = (const float*)d_in[2];   // (3D)
    const float* w_out = (const float*)d_in[3];   // (D, D)
    const float* b_out = (const float*)d_in[4];   // (D)
    const int*   mask  = (const int*)d_in[5];     // (B,S)
    float* out = (float*)d_out;

    char* ws = (char*)d_ws;
    const size_t n_x   = (size_t)Bq * Sq * Dq;            // 8M elems
    const size_t n_qkv = (size_t)Bq * Hq * Sq * HDq;      // 8M elems per tensor
    u16_t* xb   = (u16_t*)(ws);                    ws += n_x * 2;            // 16MB
    u16_t* wiT  = (u16_t*)(ws);                    ws += (size_t)3 * Dq * Dq * 2; // 6MB
    u16_t* woT  = (u16_t*)(ws);                    ws += (size_t)Dq * Dq * 2;     // 2MB
    u16_t* Qb   = (u16_t*)(ws);                    ws += n_qkv * 2;          // 16MB
    u16_t* Kb   = (u16_t*)(ws);                    ws += n_qkv * 2;          // 16MB
    u16_t* Vt   = (u16_t*)(ws);                    ws += n_qkv * 2;          // 16MB
    u16_t* attn = (u16_t*)(ws);                    /* 16MB; total ~88MB */

    // 1) conversions
    cvt_bf16_kernel<<<(int)((n_x + 255) / 256), 256, 0, stream>>>(x, xb, (int)n_x);
    transpose_bf16_kernel<<<(3 * Dq * Dq + 255) / 256, 256, 0, stream>>>(w_in, wiT, Dq, 3 * Dq);
    transpose_bf16_kernel<<<(Dq * Dq + 255) / 256, 256, 0, stream>>>(w_out, woT, Dq, Dq);

    // 2) QKV projection: 256 mtiles * 48 ngroups / 8 waves = 1536 blocks
    qkv_gemm_kernel<<<1536, 256, 0, stream>>>(xb, wiT, b_in, Qb, Kb, Vt);

    // 3) flash attention: B*H * (S/128) = 1024 blocks
    flash_attn_kernel<<<1024, 256, 0, stream>>>(Qb, Kb, Vt, mask, attn);

    // 4) output projection: 256 mtiles * 16 ngroups / 8 waves = 512 blocks
    out_gemm_kernel<<<512, 256, 0, stream>>>(attn, woT, b_out, out);
}
